// MultiHeadAttention_42580305773306
// MI455X (gfx1250) — matrix operand order repaired and checked by reading, not compiled, not run on hardware
//
#include <hip/hip_runtime.h>

#define B_  4
#define S_  1024
#define D_  1024
#define H_  16
#define DK_ 64

typedef __bf16 bf16_t;
typedef __attribute__((ext_vector_type(16))) __bf16 v16bf;
typedef __attribute__((ext_vector_type(8)))  __bf16 v8bf;
typedef __attribute__((ext_vector_type(4)))  __bf16 v4bf;
typedef __attribute__((ext_vector_type(8)))  float  v8f;
typedef __attribute__((ext_vector_type(4)))  float  v4f;

#define WMMA_BF16(A, Bm, C) \
  __builtin_amdgcn_wmma_f32_16x16x32_bf16(false, (A), false, (Bm), (short)0, (C), false, false)

// ---------------------------------------------------------------------------
// Kernel 0: fp32 -> bf16 conversion (one pass; makes GEMM loops pure-bf16)
// ---------------------------------------------------------------------------
__global__ __launch_bounds__(256)
void cvt_bf16_kernel(const float* __restrict__ src, bf16_t* __restrict__ dst, int n4)
{
  int i = blockIdx.x * 256 + threadIdx.x;
  if (i < n4) {
    v4f v = *(const v4f*)(src + (size_t)i * 4);
    v4bf o;
    #pragma unroll
    for (int k = 0; k < 4; ++k) o[k] = (bf16_t)v[k];
    *(v4bf*)(dst + (size_t)i * 4) = o;
  }
}

// ---------------------------------------------------------------------------
// Kernel 1: projection GEMM, templated on which projection (no divergence).
// One wave computes a 32x64 output block: 2 A-frags x 4 B-frags = 8 WMMA/k-step.
// P==0: Q (scaled 1/sqrt(DK)) -> [B,H,S,DK]
// P==1: K                      -> [B,H,S,DK]
// P==2: V (transposed)         -> [B,H,DK,S]
// ---------------------------------------------------------------------------
template<int P>
__global__ __launch_bounds__(128)
void proj_kernel(const bf16_t* __restrict__ X, const bf16_t* __restrict__ W,
                 const float* __restrict__ bias, bf16_t* __restrict__ Out)
{
  const int wave = threadIdx.x >> 5;
  const int lane = threadIdx.x & 31;
  const int half = lane >> 4;
  const int lr   = lane & 15;

  int tile = blockIdx.x * 4 + wave;       // 0..2047
  int m0   = (tile >> 4) * 32;            // 128 m-blocks of 32 rows
  int n0   = (tile & 15) * 64;            // 16 n-strips of 64 cols

  const bf16_t* ar[2];
  ar[0] = X + (size_t)(m0 + lr) * D_;
  ar[1] = X + (size_t)(m0 + 16 + lr) * D_;
  const bf16_t* br[4];
  #pragma unroll
  for (int j = 0; j < 4; ++j) br[j] = W + (size_t)(n0 + j * 16 + lr) * D_;

  v8f acc[2][4];
  #pragma unroll
  for (int mi = 0; mi < 2; ++mi)
    #pragma unroll
    for (int j = 0; j < 4; ++j) acc[mi][j] = 0;

  for (int k0 = 0; k0 < D_; k0 += 32) {
    __builtin_prefetch(ar[0] + k0 + 128, 0, 3);
    __builtin_prefetch(ar[1] + k0 + 128, 0, 3);
    __builtin_prefetch(br[0] + k0 + 128, 0, 3);
    v16bf A[2];
    #pragma unroll
    for (int mi = 0; mi < 2; ++mi) {
      v8bf g0 = *(const v8bf*)(ar[mi] + k0 + half * 16);
      v8bf g1 = *(const v8bf*)(ar[mi] + k0 + half * 16 + 8);
      #pragma unroll
      for (int i = 0; i < 8; ++i) { A[mi][i] = g0[i]; A[mi][8 + i] = g1[i]; }
    }
    v16bf Bf[4];
    #pragma unroll
    for (int j = 0; j < 4; ++j) Bf[j] = *(const v16bf*)(br[j] + k0 + half * 16);
    #pragma unroll
    for (int mi = 0; mi < 2; ++mi)
      #pragma unroll
      for (int j = 0; j < 4; ++j)
        acc[mi][j] = WMMA_BF16(A[mi], Bf[j], acc[mi][j]);
  }

  const float scale = (P == 0) ? 0.125f : 1.0f;   // 1/sqrt(64) folded into Q
  #pragma unroll
  for (int j = 0; j < 4; ++j) {
    int e  = n0 + j * 16 + lr;
    int hh = e >> 6;
    int dk = e & 63;
    float bv = bias[e];
    #pragma unroll
    for (int mi = 0; mi < 2; ++mi) {
      #pragma unroll
      for (int r = 0; r < 8; ++r) {
        int grow = m0 + mi * 16 + r + half * 8;
        int bb = grow >> 10;
        int ss = grow & (S_ - 1);
        float val = (acc[mi][j][r] + bv) * scale;
        if (P == 2)
          Out[((size_t)(bb * H_ + hh) * DK_ + dk) * S_ + ss] = (bf16_t)val;
        else
          Out[((size_t)(bb * H_ + hh) * S_ + ss) * DK_ + dk] = (bf16_t)val;
      }
    }
  }
}

// ---------------------------------------------------------------------------
// Kernel 2: attention. One wave owns a 16-row q-tile for one (b,h).
// 3 passes over its 16x1024 score block (L2-resident between passes).
// ---------------------------------------------------------------------------
__global__ __launch_bounds__(128)
void attn_kernel(const bf16_t* __restrict__ Qb, const bf16_t* __restrict__ Kb,
                 const bf16_t* __restrict__ Vt, const unsigned char* __restrict__ mask,
                 float* __restrict__ attn, float* __restrict__ ctx)
{
  __shared__ float smax[4][16];
  const int wave = threadIdx.x >> 5;
  const int lane = threadIdx.x & 31;
  const int half = lane >> 4;
  const int lr   = lane & 15;

  int gw  = blockIdx.x * 4 + wave;           // 0..4095
  int b   = gw >> 10;
  int rem = gw & 1023;
  int h   = rem >> 6;
  int q0  = (rem & 63) << 4;
  const int bh = b * H_ + h;

  const bf16_t* Qp = Qb + (size_t)bh * S_ * DK_;
  const bf16_t* Kp = Kb + (size_t)bh * S_ * DK_;
  const bf16_t* Vp = Vt + (size_t)bh * DK_ * S_;
  float* arow = attn + ((size_t)bh * S_ + q0) * S_;
  const unsigned char* mrow = mask + ((size_t)b * S_ + q0) * S_;

  // Q A-fragments for d-chunks [0,32) and [32,64): loaded once.
  v16bf AQ[2];
  {
    const bf16_t* qr = Qp + (size_t)(q0 + lr) * DK_;
    #pragma unroll
    for (int c = 0; c < 2; ++c) {
      v8bf g0 = *(const v8bf*)(qr + c * 32 + half * 16);
      v8bf g1 = *(const v8bf*)(qr + c * 32 + half * 16 + 8);
      #pragma unroll
      for (int i = 0; i < 8; ++i) { AQ[c][i] = g0[i]; AQ[c][8 + i] = g1[i]; }
    }
  }

  // ---- Phase 1: scores = (Q/sqrt(dk)) K^T, mask, write raw, track row max
  float rm[8];
  #pragma unroll
  for (int r = 0; r < 8; ++r) rm[r] = -3.0e38f;

  for (int k0 = 0; k0 < S_; k0 += 16) {
    const bf16_t* kr = Kp + (size_t)(k0 + lr) * DK_;
    __builtin_prefetch(kr + 16 * DK_, 0, 3);            // next k-tile row
    v16bf B0 = *(const v16bf*)(kr + half * 16);         // d = 0..31
    v16bf B1 = *(const v16bf*)(kr + 32 + half * 16);    // d = 32..63
    v8f c8 = 0;
    c8 = WMMA_BF16(AQ[0], B0, c8);
    c8 = WMMA_BF16(AQ[1], B1, c8);
    #pragma unroll
    for (int r = 0; r < 8; ++r) {
      int m = r + half * 8;
      float s = mrow[(size_t)m * S_ + k0 + lr] ? -1.0e9f : c8[r];
      rm[r] = fmaxf(rm[r], s);
      arow[(size_t)m * S_ + k0 + lr] = s;
    }
  }
  // reduce row max across the 16-lane half that owns each row
  #pragma unroll
  for (int r = 0; r < 8; ++r) {
    #pragma unroll
    for (int off = 8; off >= 1; off >>= 1)
      rm[r] = fmaxf(rm[r], __shfl_xor(rm[r], off, 32));
  }
  if (lane == 0 || lane == 16) {
    #pragma unroll
    for (int r = 0; r < 8; ++r) smax[wave][half * 8 + r] = rm[r];
  }
  __syncthreads();
  const float qmax = smax[wave][lr];      // max for row (q0 + lr)

  // ---- Phase 2a: row sums of exp(s - max)
  const float* prow = arow + (size_t)lr * S_;
  float lsum = 0.f;
  for (int kc = 0; kc < S_; kc += 32) {
    v4f s0 = *(const v4f*)(prow + kc + half * 8);
    v4f s1 = *(const v4f*)(prow + kc + half * 8 + 4);
    v4f s2 = *(const v4f*)(prow + kc + 16 + half * 8);
    v4f s3 = *(const v4f*)(prow + kc + 16 + half * 8 + 4);
    #pragma unroll
    for (int i = 0; i < 4; ++i)
      lsum += __expf(s0[i] - qmax) + __expf(s1[i] - qmax) +
              __expf(s2[i] - qmax) + __expf(s3[i] - qmax);
  }
  lsum += __shfl_xor(lsum, 16, 32);
  const float inv = 1.0f / lsum;

  // ---- Phase 2b: write normalized probs + context = P * V
  float* wrow = arow + (size_t)lr * S_;
  v8f cacc[4];
  #pragma unroll
  for (int i = 0; i < 4; ++i) cacc[i] = 0;

  for (int kc = 0; kc < S_; kc += 32) {
    v4f s0 = *(const v4f*)(prow + kc + half * 8);
    v4f s1 = *(const v4f*)(prow + kc + half * 8 + 4);
    v4f s2 = *(const v4f*)(prow + kc + 16 + half * 8);
    v4f s3 = *(const v4f*)(prow + kc + 16 + half * 8 + 4);
    v4f p0, p1, p2, p3;
    #pragma unroll
    for (int i = 0; i < 4; ++i) {
      p0[i] = __expf(s0[i] - qmax) * inv;
      p1[i] = __expf(s1[i] - qmax) * inv;
      p2[i] = __expf(s2[i] - qmax) * inv;
      p3[i] = __expf(s3[i] - qmax) * inv;
    }
    *(v4f*)(wrow + kc + half * 8)          = p0;
    *(v4f*)(wrow + kc + half * 8 + 4)      = p1;
    *(v4f*)(wrow + kc + 16 + half * 8)     = p2;
    *(v4f*)(wrow + kc + 16 + half * 8 + 4) = p3;
    v16bf A;
    #pragma unroll
    for (int i = 0; i < 4; ++i) {
      A[i]      = (bf16_t)p0[i];  A[4 + i]  = (bf16_t)p1[i];
      A[8 + i]  = (bf16_t)p2[i];  A[12 + i] = (bf16_t)p3[i];
    }
    #pragma unroll
    for (int dt = 0; dt < 4; ++dt) {
      const bf16_t* vr = Vp + (size_t)(dt * 16 + lr) * S_ + kc;
      v8bf v0 = *(const v8bf*)(vr + half * 8);
      v8bf v1 = *(const v8bf*)(vr + 16 + half * 8);
      v16bf Bv;
      #pragma unroll
      for (int i = 0; i < 8; ++i) { Bv[i] = v0[i]; Bv[8 + i] = v1[i]; }
      cacc[dt] = WMMA_BF16(A, Bv, cacc[dt]);
    }
  }
  // store context tile -> ctx [B,S,D]
  #pragma unroll
  for (int dt = 0; dt < 4; ++dt) {
    #pragma unroll
    for (int r = 0; r < 8; ++r) {
      int m = r + half * 8;
      int d = h * DK_ + dt * 16 + lr;
      ctx[((size_t)b * S_ + q0 + m) * D_ + d] = cacc[dt][r];
    }
  }
}

// ---------------------------------------------------------------------------
// Kernel 3: residual + LayerNorm over D.  One block per (b,s) row.
// ---------------------------------------------------------------------------
__global__ __launch_bounds__(256)
void ln_kernel(const float* __restrict__ query, const float* __restrict__ ctx,
               const float* __restrict__ gamma, const float* __restrict__ beta,
               float* __restrict__ out)
{
  __shared__ float ssum[8], ssq[8];
  const int row = blockIdx.x;
  const float* qrow = query + (size_t)row * D_;
  const float* crow = ctx + (size_t)row * D_;

  float x[4], s = 0.f, sq = 0.f;
  #pragma unroll
  for (int t = 0; t < 4; ++t) {
    int i = threadIdx.x + t * 256;
    x[t] = qrow[i] + crow[i];
    s += x[t]; sq += x[t] * x[t];
  }
  #pragma unroll
  for (int off = 16; off >= 1; off >>= 1) {
    s  += __shfl_xor(s,  off, 32);
    sq += __shfl_xor(sq, off, 32);
  }
  int wv = threadIdx.x >> 5;
  if ((threadIdx.x & 31) == 0) { ssum[wv] = s; ssq[wv] = sq; }
  __syncthreads();
  float ts = 0.f, tq = 0.f;
  #pragma unroll
  for (int i = 0; i < 8; ++i) { ts += ssum[i]; tq += ssq[i]; }
  const float mu  = ts * (1.0f / D_);
  const float var = tq * (1.0f / D_) - mu * mu;
  const float rs  = rsqrtf(var + 1e-5f);
  #pragma unroll
  for (int t = 0; t < 4; ++t) {
    int i = threadIdx.x + t * 256;
    out[(size_t)row * D_ + i] = (x[t] - mu) * rs * gamma[i] + beta[i];
  }
}

// ---------------------------------------------------------------------------
extern "C" void kernel_launch(void* const* d_in, const int* in_sizes, int n_in,
                              void* d_out, int out_size, void* d_ws, size_t ws_size,
                              hipStream_t stream) {
  const float* query = (const float*)d_in[0];
  const float* key_  = (const float*)d_in[1];
  const float* value = (const float*)d_in[2];
  const unsigned char* mask = (const unsigned char*)d_in[3];  // numpy bool bytes
  const float* Wq = (const float*)d_in[4];  const float* bq = (const float*)d_in[5];
  const float* Wk = (const float*)d_in[6];  const float* bk = (const float*)d_in[7];
  const float* Wv = (const float*)d_in[8];  const float* bv = (const float*)d_in[9];
  const float* gamma = (const float*)d_in[10];
  const float* beta  = (const float*)d_in[11];

  float* out  = (float*)d_out;                          // [B,S,D]
  float* attn = out + (size_t)B_ * S_ * D_;             // [B,H,S,S]

  const size_t MB = 1024 * 1024;
  char* ws = (char*)d_ws;
  bf16_t* Xqb = (bf16_t*)(ws + 0 * MB);    // 8 MB  bf16 [B*S, D]
  bf16_t* Xkb = (bf16_t*)(ws + 8 * MB);    // 8 MB
  bf16_t* Xvb = (bf16_t*)(ws + 16 * MB);   // 8 MB
  bf16_t* Wqb = (bf16_t*)(ws + 24 * MB);   // 2 MB  bf16 [H*DK, D]
  bf16_t* Wkb = (bf16_t*)(ws + 26 * MB);   // 2 MB
  bf16_t* Wvb = (bf16_t*)(ws + 28 * MB);   // 2 MB
  bf16_t* Qb  = (bf16_t*)(ws + 32 * MB);   // 8 MB  [B,H,S,DK]
  bf16_t* Kb  = (bf16_t*)(ws + 40 * MB);   // 8 MB  [B,H,S,DK]
  bf16_t* Vt  = (bf16_t*)(ws + 48 * MB);   // 8 MB  [B,H,DK,S]
  float*  ctx = (float*)(ws + 56 * MB);    // 16 MB fp32 [B,S,D]

  const int NX4 = B_ * S_ * D_ / 4;        // 1,048,576 vec4
  const int NW4 = H_ * DK_ * D_ / 4;       //   262,144 vec4
  cvt_bf16_kernel<<<NX4 / 256, 256, 0, stream>>>(query, Xqb, NX4);
  cvt_bf16_kernel<<<NX4 / 256, 256, 0, stream>>>(key_,  Xkb, NX4);
  cvt_bf16_kernel<<<NX4 / 256, 256, 0, stream>>>(value, Xvb, NX4);
  cvt_bf16_kernel<<<NW4 / 256, 256, 0, stream>>>(Wq, Wqb, NW4);
  cvt_bf16_kernel<<<NW4 / 256, 256, 0, stream>>>(Wk, Wkb, NW4);
  cvt_bf16_kernel<<<NW4 / 256, 256, 0, stream>>>(Wv, Wvb, NW4);

  // 2048 waves (32x64 blocks) per projection, 4 waves/block
  proj_kernel<0><<<512, 128, 0, stream>>>(Xqb, Wqb, bq, Qb);
  proj_kernel<1><<<512, 128, 0, stream>>>(Xkb, Wkb, bk, Kb);
  proj_kernel<2><<<512, 128, 0, stream>>>(Xvb, Wvb, bv, Vt);

  // 4096 q-tiles, 4 waves/block
  attn_kernel<<<4096 / 4, 128, 0, stream>>>(Qb, Kb, Vt, mask, attn, ctx);

  // one block per (b,s) row
  ln_kernel<<<B_ * S_, 256, 0, stream>>>(query, ctx, gamma, beta, out);
}
